// MambaLayer_3728031613109
// MI455X (gfx1250) — matrix operand order repaired and checked by reading, not compiled, not run on hardware
//
#include <hip/hip_runtime.h>
#include <hip/hip_bf16.h>
#include <cstdint>

typedef __bf16 bf16;
typedef __attribute__((ext_vector_type(2)))  __bf16 v2bf;
typedef __attribute__((ext_vector_type(4)))  __bf16 v4bf;
typedef __attribute__((ext_vector_type(8)))  __bf16 v8bf;
typedef __attribute__((ext_vector_type(16))) __bf16 v16bf;
typedef __attribute__((ext_vector_type(8)))  float  v8f;

// ---- optional CDNA5 TDM path (compile-safe guards) ----
#if defined(__has_builtin)
#  if __has_builtin(__builtin_amdgcn_tensor_load_to_lds)
#    define HAVE_TDM 1
#  else
#    define HAVE_TDM 0
#  endif
#else
#  define HAVE_TDM 0
#endif

#if defined(__has_include)
#  if __has_include(<hip/amd_detail/amd_gfx1250_TDM.h>)
#    define TDM_6ARG 1
#  else
#    define TDM_6ARG 0
#  endif
#else
#  define TDM_6ARG 0
#endif

static constexpr int kDM    = 1024;           // D_MODEL
static constexpr int kDS    = 16;             // D_STATE
static constexpr int kDI    = 2048;           // D_INNER
static constexpr int kN1    = 2 * kDI;        // 4096 (W_in cols)
static constexpr int kB     = 4;
static constexpr int kL     = 4096;
static constexpr int kM     = kB * kL;        // 16384 rows

// ---- workspace layout (bytes) ----
static constexpr size_t OFF_XB    = 0;                                    // bf16 x        (M x DM)
static constexpr size_t OFF_WINB  = OFF_XB    + (size_t)kM  * kDM * 2;    // bf16 W_in     (DM x N1)
static constexpr size_t OFF_WOUTB = OFF_WINB  + (size_t)kDM * kN1 * 2;    // bf16 W_out    (DI x DM)
static constexpr size_t OFF_XMAIN = OFF_WOUTB + (size_t)kDI * kDM * 2;    // f32 x_main    (M x DI)
static constexpr size_t OFF_RES   = OFF_XMAIN + (size_t)kM  * kDI * 4;    // f32 res       (M x DI)
static constexpr size_t OFF_XC    = OFF_RES   + (size_t)kM  * kDI * 4;    // f32 silu(conv)(M x DI)
static constexpr size_t OFF_BT    = OFF_XC    + (size_t)kM  * kDI * 4;    // f32 Bt        (M x 16)
static constexpr size_t OFF_CT    = OFF_BT    + (size_t)kM  * kDS * 4;    // f32 Ct        (M x 16)
static constexpr size_t OFF_YS    = OFF_CT    + (size_t)kM  * kDS * 4;    // f32 ys        (M)

__device__ __forceinline__ float silu_f(float v) {
    return v / (1.0f + __expf(-v));
}

#if HAVE_TDM
typedef __attribute__((ext_vector_type(4))) unsigned int v4u;
typedef __attribute__((ext_vector_type(4))) int          v4i;
typedef __attribute__((ext_vector_type(8))) int          v8i_;

// 2D tile load: rows x k_elems (bf16), global row stride = row_stride_elems.
// Descriptor packing per CDNA5 ISA 8.3/8.4.
__device__ __forceinline__ void tdm_load_tile_2d(const void* gptr, unsigned lds_addr,
                                                 unsigned k_elems, unsigned rows,
                                                 unsigned row_stride_elems) {
    unsigned long long ga = (unsigned long long)(uintptr_t)gptr;
    v4u g0 = { 1u,                                   // count=1, user mode
               lds_addr,                             // lds_addr[31:0]
               (unsigned)ga,                         // global_addr[31:0]
               (unsigned)((ga >> 32) & 0x01FFFFFFu) | (2u << 30) };  // ga[56:32], type=2
    v8i_ g1 = { (int)(1u << 16),                                     // data_size=1 (2 bytes)
                (int)((k_elems & 0xFFFFu) << 16),                    // tensor_dim0 lo
                (int)(((k_elems >> 16) & 0xFFFFu) | ((rows & 0xFFFFu) << 16)),   // td0 hi | td1 lo
                (int)(((rows >> 16) & 0xFFFFu) | ((k_elems & 0xFFFFu) << 16)),   // td1 hi | tile_dim0
                (int)(rows & 0xFFFFu),                               // tile_dim1 (tile_dim2=0)
                (int)row_stride_elems,                               // tensor_dim0_stride[31:0]
                0, 0 };
    v4i z4 = { 0, 0, 0, 0 };
#if TDM_6ARG
    v8i_ z8 = { 0, 0, 0, 0, 0, 0, 0, 0 };
    __builtin_amdgcn_tensor_load_to_lds(g0, g1, z4, z4, z8, 0);
#else
    __builtin_amdgcn_tensor_load_to_lds(g0, g1, z4, z4, 0);
#endif
}

__device__ __forceinline__ void tdm_wait0() {
#if defined(__has_builtin) && __has_builtin(__builtin_amdgcn_s_wait_tensorcnt)
    __builtin_amdgcn_s_wait_tensorcnt(0);
#else
    asm volatile("s_wait_tensorcnt 0x0" ::: "memory");
#endif
}
#endif  // HAVE_TDM

// ------------------------------------------------------------------
// K0: fp32 -> bf16 conversion
// ------------------------------------------------------------------
__global__ __launch_bounds__(256) void cvt_f32_bf16(const float* __restrict__ src,
                                                    bf16* __restrict__ dst, int n) {
    int i = (blockIdx.x * 256 + threadIdx.x) * 4;
    if (i + 3 < n) {
        float4 v = *(const float4*)(src + i);
        dst[i + 0] = (bf16)v.x;
        dst[i + 1] = (bf16)v.y;
        dst[i + 2] = (bf16)v.z;
        dst[i + 3] = (bf16)v.w;
    } else {
        for (; i < n; ++i) dst[i] = (bf16)src[i];
    }
}

// ------------------------------------------------------------------
// K1: GEMM1  xr = x(bf16, MxDM) @ W_in(bf16, DMxN1), split-> x_main / res
// Double-buffered LDS; A tile staged by the Tensor Data Mover (wave 0),
// overlapped with WMMA compute of the previous tile.
// ------------------------------------------------------------------
__global__ __launch_bounds__(256) void gemm_in_wmma(const bf16* __restrict__ A,
                                                    const bf16* __restrict__ Bm,
                                                    float* __restrict__ Xmain,
                                                    float* __restrict__ Res) {
    __shared__ __align__(32) bf16 As[2][128][32];   // [buf][m][k]
    __shared__ __align__(32) bf16 Bs[2][128][32];   // [buf][n][k]

    const int K = kDM, ldb = kN1;
    const int bm = blockIdx.y * 128, bn = blockIdx.x * 128;
    const int tid = threadIdx.x;
    const int wid = tid >> 5, lane = tid & 31;
    const int wm = (wid & 3) * 32, wn = (wid >> 2) * 64;
    const int l16 = lane & 15;
    const int kA = (lane >> 4) * 16;   // A K-half base
    const int kBf = (lane >> 4) * 16;  // B K-run base

    auto stage_tiles = [&](int buf, int kk0) {
#if HAVE_TDM
        if (wid == 0)
            tdm_load_tile_2d(A + (size_t)bm * K + kk0,
                             (unsigned)(uintptr_t)&As[buf][0][0], 32u, 128u, (unsigned)K);
#else
        {   // stage A tile: 16 bf16 per thread
            int f = tid * 16;
            int r = f >> 5, c = f & 31;
            const bf16* g = A + (size_t)(bm + r) * K + kk0 + c;
            *(v8bf*)&As[buf][r][c]     = *(const v8bf*)g;
            *(v8bf*)&As[buf][r][c + 8] = *(const v8bf*)(g + 8);
        }
#endif
        {   // stage B tile (transpose -> [n][k]); two K-rows per thread, dword stores
            int kk2 = (tid >> 4) * 2;       // 0..30 step 2
            int nn  = (tid & 15) * 8;       // 0..120
            const bf16* gp = Bm + (size_t)(kk0 + kk2) * ldb + bn + nn;
            v8bf r0 = *(const v8bf*)gp;
            v8bf r1 = *(const v8bf*)(gp + ldb);
#pragma unroll
            for (int j = 0; j < 8; ++j) {
                v2bf pr = { r0[j], r1[j] };
                *(v2bf*)&Bs[buf][nn + j][kk2] = pr;
            }
        }
    };

    v8f acc[2][4];
#pragma unroll
    for (int mt = 0; mt < 2; ++mt)
#pragma unroll
        for (int nt = 0; nt < 4; ++nt) acc[mt][nt] = (v8f){0.f,0.f,0.f,0.f,0.f,0.f,0.f,0.f};

    // prologue: fill buffer 0
    stage_tiles(0, 0);
#if HAVE_TDM
    if (wid == 0) tdm_wait0();
#endif
    __syncthreads();

    int buf = 0;
    for (int k0 = 0; k0 < K; k0 += 32, buf ^= 1) {
        const int nk = k0 + 32;
        if (nk < K) {
            stage_tiles(buf ^ 1, nk);           // overlap with compute below
            if (nk + 32 < K) {                  // prefetch the tile after that
                __builtin_prefetch(A + (size_t)(bm + (tid >> 1)) * K + (nk + 32), 0, 0);
                __builtin_prefetch(Bm + (size_t)(nk + 32 + (tid >> 4)) * ldb + bn, 0, 0);
            }
        }

        v16bf af[2], bfrag[4];
#pragma unroll
        for (int mt = 0; mt < 2; ++mt) {
            int m = wm + mt * 16 + l16;
            v8bf lo = *(const v8bf*)&As[buf][m][kA];
            v8bf hi = *(const v8bf*)&As[buf][m][kA + 8];
#pragma unroll
            for (int j = 0; j < 8; ++j) { af[mt][j] = lo[j]; af[mt][8 + j] = hi[j]; }
        }
#pragma unroll
        for (int nt = 0; nt < 4; ++nt) {
            int n = wn + nt * 16 + l16;
            bfrag[nt] = *(const v16bf*)&Bs[buf][n][kBf];
        }
#pragma unroll
        for (int mt = 0; mt < 2; ++mt)
#pragma unroll
            for (int nt = 0; nt < 4; ++nt)
                acc[mt][nt] = __builtin_amdgcn_wmma_f32_16x16x32_bf16(
                    false, af[mt], false, bfrag[nt], (short)0, acc[mt][nt], false, false);

#if HAVE_TDM
        if (nk < K && wid == 0) tdm_wait0();
#endif
        __syncthreads();   // covers: next buffer written AND current buffer consumed
    }

    // epilogue: uniform split into x_main / res
    const int n_l = lane & 15, mhi = (lane >> 4) * 8;
    float* dbase;
    int cb;
    if (bn < kDI) { dbase = Xmain; cb = bn; }
    else          { dbase = Res;   cb = bn - kDI; }
#pragma unroll
    for (int mt = 0; mt < 2; ++mt)
#pragma unroll
        for (int nt = 0; nt < 4; ++nt) {
            float* p = dbase + (size_t)(bm + wm + mt * 16 + mhi) * kDI
                             + (size_t)(cb + wn + nt * 16 + n_l);
#pragma unroll
            for (int r = 0; r < 8; ++r) p[(size_t)r * kDI] = acc[mt][nt][r];
        }
}

// ------------------------------------------------------------------
// K2: depthwise conv(k=3, pad 1) + bias + SiLU
// ------------------------------------------------------------------
__global__ __launch_bounds__(256) void conv_silu(const float* __restrict__ xm,
                                                 const float* __restrict__ cw,
                                                 const float* __restrict__ cb,
                                                 float* __restrict__ xc) {
    size_t idx = (size_t)blockIdx.x * 256 + threadIdx.x;  // over M*DI
    int i = (int)(idx % kDI);
    size_t row = idx / kDI;
    int l = (int)(row % kL);
    float w0 = cw[i * 3 + 0], w1 = cw[i * 3 + 1], w2 = cw[i * 3 + 2];
    float a = cb[i] + w1 * xm[idx];
    if (l > 0)      a += w0 * xm[idx - kDI];
    if (l < kL - 1) a += w2 * xm[idx + kDI];
    xc[idx] = silu_f(a);
}

// ------------------------------------------------------------------
// K3: Bt/Ct projections (per-row reduction over DI, N=16 each)
// ------------------------------------------------------------------
__global__ __launch_bounds__(256) void proj_bc(const float* __restrict__ xc,
                                               const float* __restrict__ WB,
                                               const float* __restrict__ WC,
                                               float* __restrict__ Bt,
                                               float* __restrict__ Ct) {
    __shared__ float red[2][16];
    const int row = blockIdx.x, tid = threadIdx.x;
    if (tid < 32) red[tid >> 4][tid & 15] = 0.f;
    __syncthreads();
    float aB[16], aC[16];
#pragma unroll
    for (int s = 0; s < 16; ++s) { aB[s] = 0.f; aC[s] = 0.f; }
    const float* xr = xc + (size_t)row * kDI;
    for (int i = tid; i < kDI; i += 256) {
        float xv = xr[i];
        const float* wb = WB + i * 16;
        const float* wc = WC + i * 16;
#pragma unroll
        for (int s = 0; s < 16; ++s) { aB[s] += xv * wb[s]; aC[s] += xv * wc[s]; }
    }
#pragma unroll
    for (int s = 0; s < 16; ++s) {
        atomicAdd(&red[0][s], aB[s]);
        atomicAdd(&red[1][s], aC[s]);
    }
    __syncthreads();
    if (tid < 16)       Bt[(size_t)row * 16 + tid]        = red[0][tid];
    else if (tid < 32)  Ct[(size_t)row * 16 + (tid - 16)] = red[1][tid - 16];
}

// ------------------------------------------------------------------
// K4: sequential state-space scan. One wave per batch; lanes 0..15 = state.
// ------------------------------------------------------------------
__global__ void scan_kernel(const float* __restrict__ Bt, const float* __restrict__ Ct,
                            const float* __restrict__ A, float* __restrict__ ys) {
    const int b = blockIdx.x;
    const int lane = threadIdx.x;   // 32 threads
    const int s = lane & 15;
    float a = A[s];
    float sp = (a > 20.f) ? a : __logf(1.f + __expf(a));
    float decay = __expf(-sp);
    float state = 0.f;
    const float* Bp = Bt + (size_t)b * kL * 16;
    const float* Cp = Ct + (size_t)b * kL * 16;
    for (int t = 0; t < kL; ++t) {
        float bv = Bp[t * 16 + s];
        float cv = Cp[t * 16 + s];
        state = state * decay + bv;
        float prod = (lane < 16) ? state * cv : 0.f;
        for (int off = 8; off > 0; off >>= 1) prod += __shfl_down(prod, off, 32);
        if (lane == 0) ys[(size_t)b * kL + t] = prod;
    }
}

// ------------------------------------------------------------------
// K5: GEMM2  out = y @ W_out;  y built on the fly:
//     y[m,i] = (ys[m] + xc[m,i]*D[i]) * silu(res[m,i])  -> bf16 -> LDS
// Double-buffered LDS.
// ------------------------------------------------------------------
__global__ __launch_bounds__(256) void gemm_out_wmma(const float* __restrict__ xc,
                                                     const float* __restrict__ resv,
                                                     const float* __restrict__ ys,
                                                     const float* __restrict__ Dv,
                                                     const bf16* __restrict__ Bm,
                                                     float* __restrict__ out) {
    __shared__ __align__(32) bf16 As[2][128][32];
    __shared__ __align__(32) bf16 Bs[2][128][32];

    const int K = kDI, ldb = kDM;
    const int bm = blockIdx.y * 128, bn = blockIdx.x * 128;
    const int tid = threadIdx.x;
    const int wid = tid >> 5, lane = tid & 31;
    const int wm = (wid & 3) * 32, wn = (wid >> 2) * 64;
    const int l16 = lane & 15;
    const int kA = (lane >> 4) * 16;
    const int kBf = (lane >> 4) * 16;

    auto stage_tiles = [&](int buf, int kk0) {
        {   // A tile: fused elementwise state-path epilogue (b64 LDS stores)
            int f = tid * 16;
            int r = f >> 5, c = f & 31;
            size_t row = (size_t)(bm + r);
            float yv = ys[row];
#pragma unroll
            for (int g = 0; g < 4; ++g) {
                int kk = c + g * 4;
                size_t gi = row * (size_t)kDI + (size_t)(kk0 + kk);
                float4 xv = *(const float4*)(xc + gi);
                float4 rv = *(const float4*)(resv + gi);
                float4 dv = *(const float4*)(Dv + kk0 + kk);
                v4bf pk = { (bf16)((yv + xv.x * dv.x) * silu_f(rv.x)),
                            (bf16)((yv + xv.y * dv.y) * silu_f(rv.y)),
                            (bf16)((yv + xv.z * dv.z) * silu_f(rv.z)),
                            (bf16)((yv + xv.w * dv.w) * silu_f(rv.w)) };
                *(v4bf*)&As[buf][r][kk] = pk;
            }
        }
        {   // B tile (transpose -> [n][k]); dword stores
            int kk2 = (tid >> 4) * 2;
            int nn  = (tid & 15) * 8;
            const bf16* gp = Bm + (size_t)(kk0 + kk2) * ldb + bn + nn;
            v8bf r0 = *(const v8bf*)gp;
            v8bf r1 = *(const v8bf*)(gp + ldb);
#pragma unroll
            for (int j = 0; j < 8; ++j) {
                v2bf pr = { r0[j], r1[j] };
                *(v2bf*)&Bs[buf][nn + j][kk2] = pr;
            }
        }
    };

    v8f acc[2][4];
#pragma unroll
    for (int mt = 0; mt < 2; ++mt)
#pragma unroll
        for (int nt = 0; nt < 4; ++nt) acc[mt][nt] = (v8f){0.f,0.f,0.f,0.f,0.f,0.f,0.f,0.f};

    stage_tiles(0, 0);
    __syncthreads();

    int buf = 0;
    for (int k0 = 0; k0 < K; k0 += 32, buf ^= 1) {
        const int nk = k0 + 32;
        if (nk < K) {
            stage_tiles(buf ^ 1, nk);
            if (nk + 32 < K) {
                __builtin_prefetch(xc + (size_t)(bm + (tid >> 1)) * kDI + (nk + 32), 0, 0);
                __builtin_prefetch(Bm + (size_t)(nk + 32 + (tid >> 4)) * ldb + bn, 0, 0);
            }
        }

        v16bf af[2], bfrag[4];
#pragma unroll
        for (int mt = 0; mt < 2; ++mt) {
            int m = wm + mt * 16 + l16;
            v8bf lo = *(const v8bf*)&As[buf][m][kA];
            v8bf hi = *(const v8bf*)&As[buf][m][kA + 8];
#pragma unroll
            for (int j = 0; j < 8; ++j) { af[mt][j] = lo[j]; af[mt][8 + j] = hi[j]; }
        }
#pragma unroll
        for (int nt = 0; nt < 4; ++nt) {
            int n = wn + nt * 16 + l16;
            bfrag[nt] = *(const v16bf*)&Bs[buf][n][kBf];
        }
#pragma unroll
        for (int mt = 0; mt < 2; ++mt)
#pragma unroll
            for (int nt = 0; nt < 4; ++nt)
                acc[mt][nt] = __builtin_amdgcn_wmma_f32_16x16x32_bf16(
                    false, af[mt], false, bfrag[nt], (short)0, acc[mt][nt], false, false);

        __syncthreads();
    }

    const int n_l = lane & 15, mhi = (lane >> 4) * 8;
#pragma unroll
    for (int mt = 0; mt < 2; ++mt)
#pragma unroll
        for (int nt = 0; nt < 4; ++nt) {
            float* p = out + (size_t)(bm + wm + mt * 16 + mhi) * kDM
                           + (size_t)(bn + wn + nt * 16 + n_l);
#pragma unroll
            for (int r = 0; r < 8; ++r) p[(size_t)r * kDM] = acc[mt][nt][r];
        }
}

// ------------------------------------------------------------------
extern "C" void kernel_launch(void* const* d_in, const int* in_sizes, int n_in,
                              void* d_out, int out_size, void* d_ws, size_t ws_size,
                              hipStream_t stream) {
    const float* x      = (const float*)d_in[0];
    const float* W_in   = (const float*)d_in[1];
    const float* conv_w = (const float*)d_in[2];
    const float* conv_b = (const float*)d_in[3];
    const float* W_B    = (const float*)d_in[4];
    const float* W_C    = (const float*)d_in[5];
    const float* A      = (const float*)d_in[6];
    const float* Dv     = (const float*)d_in[7];
    const float* W_out  = (const float*)d_in[8];
    float* out = (float*)d_out;

    char* ws = (char*)d_ws;
    bf16*  xb     = (bf16*)(ws + OFF_XB);
    bf16*  winb   = (bf16*)(ws + OFF_WINB);
    bf16*  woutb  = (bf16*)(ws + OFF_WOUTB);
    float* xmain  = (float*)(ws + OFF_XMAIN);
    float* resv   = (float*)(ws + OFF_RES);
    float* xcv    = (float*)(ws + OFF_XC);
    float* Btv    = (float*)(ws + OFF_BT);
    float* Ctv    = (float*)(ws + OFF_CT);
    float* ysv    = (float*)(ws + OFF_YS);

    // K0: conversions
    {
        int n0 = kM * kDM;
        int n1 = kDM * kN1;
        int n2 = kDI * kDM;
        cvt_f32_bf16<<<(n0 / 4 + 255) / 256, 256, 0, stream>>>(x, xb, n0);
        cvt_f32_bf16<<<(n1 / 4 + 255) / 256, 256, 0, stream>>>(W_in, winb, n1);
        cvt_f32_bf16<<<(n2 / 4 + 255) / 256, 256, 0, stream>>>(W_out, woutb, n2);
    }

    // K1: GEMM1 + split
    {
        dim3 grid(kN1 / 128, kM / 128);   // (32, 128)
        gemm_in_wmma<<<grid, 256, 0, stream>>>(xb, winb, xmain, resv);
    }

    // K2: conv + SiLU
    {
        size_t n = (size_t)kM * kDI;
        conv_silu<<<(unsigned)(n / 256), 256, 0, stream>>>(xmain, conv_w, conv_b, xcv);
    }

    // K3: Bt / Ct
    proj_bc<<<kM, 256, 0, stream>>>(xcv, W_B, W_C, Btv, Ctv);

    // K4: scan
    scan_kernel<<<kB, 32, 0, stream>>>(Btv, Ctv, A, ysv);

    // K5: fused elementwise + GEMM2
    {
        dim3 grid(kDM / 128, kM / 128);   // (8, 128)
        gemm_out_wmma<<<grid, 256, 0, stream>>>(xcv, resv, ysv, Dv, woutb, out);
    }
}
